// AttnDecoderRNN_61400852463706
// MI455X (gfx1250) — matrix operand order repaired and checked by reading, not compiled, not run on hardware
//
#include <hip/hip_runtime.h>

// ---------------------------------------------------------------------------
// Problem sizes (fixed by the reference)
// ---------------------------------------------------------------------------
#define H   1024
#define P   3
#define S   256
#define B   256
#define SB  (S * B)          // 65536 rows of the big GEMM
#define EPS 1e-5f

// d_out layout (float32, concatenated in reference return order)
#define OUT_PALETTE 0
#define OUT_CONTEXT 768                       // context[None] -> B*H
#define OUT_HIDDEN  (768 + B * H)             // gru_hidden    -> B*H
#define OUT_ATTN    (768 + 2 * B * H)         // attn_weights  -> B*S

// Workspace layout (bytes)
#define WS_WE      0                           // bf16 attn_e_w, H*H*2 = 2 MB
#define WS_HPROJ   (H * H * 2)                 // f32 [B,H] = 1 MB
#define WS_ENERGY  (WS_HPROJ + B * H * 4)      // f32 [S*B] = 256 KB
#define WS_WSOFT   (WS_ENERGY + SB * 4)        // f32 [S*B] = 256 KB
#define WS_Y       (WS_WSOFT + SB * 4)         // f32 [B,H] = 1 MB
#define WS_BN      (WS_Y + B * H * 4)          // f32 [2,H] = 8 KB

// LDS layout of the WMMA kernel (dynamic shared memory, base assumed 0)
#define LDS_A_BYTES (128 * H * 2)              // 256 KB A slab (bf16)
#define LDS_W_BYTES (16 * H * 2)               // 32 KB per W tile (bf16)
#define LDS_TOTAL   (LDS_A_BYTES + 2 * LDS_W_BYTES)   // 320 KB = WGP cap

// ---------------------------------------------------------------------------
// CDNA5 WMMA / TDM types
// ---------------------------------------------------------------------------
typedef __attribute__((ext_vector_type(16))) __bf16        v16bf;
typedef __attribute__((ext_vector_type(8)))  float         v8f;
typedef __attribute__((ext_vector_type(4)))  unsigned int  v4u;
typedef __attribute__((ext_vector_type(8)))  unsigned int  v8u;
typedef __attribute__((ext_vector_type(8)))  int           v8i;
typedef __attribute__((ext_vector_type(4)))  int           v4i;

__device__ __forceinline__ unsigned int pack_bf16(float lo, float hi) {
    __bf16 a = (__bf16)lo, b = (__bf16)hi;
    unsigned short ua = __builtin_bit_cast(unsigned short, a);
    unsigned short ub = __builtin_bit_cast(unsigned short, b);
    return (unsigned int)ua | ((unsigned int)ub << 16);
}

__device__ __forceinline__ float sigf(float x) { return 1.0f / (1.0f + __expf(-x)); }

#if __has_builtin(__builtin_amdgcn_tensor_load_to_lds) && \
    __has_builtin(__builtin_amdgcn_s_wait_tensorcnt)
#define HAS_TDM 1
#else
#define HAS_TDM 0
#endif

#if HAS_TDM
// Issue a 1D TDM transfer: 32 KB (one 16xH bf16 W tile) global -> LDS.
// D# per CDNA5 ISA: group0 = {count/flags, lds_addr,
// global_addr, type=2}; group1 = {data_size=8B, tensor_dim0=tile_dim0=4096,
// tensor_dim1=tile_dim1=1, dim0_stride=4096}; groups 2/3 zero (<=2D tensor).
// This toolchain exposes the 6-arg builtin (extra int32x8 descriptor slot).
__device__ __forceinline__ void tdm_load_w(const unsigned int* gsrc, unsigned int lds_off) {
    unsigned long long ga = (unsigned long long)(__SIZE_TYPE__)gsrc;
    v4u g0;
    g0[0] = 1u;                                            // count=1 (valid), flags 0
    g0[1] = lds_off;                                       // LDS byte address
    g0[2] = (unsigned int)ga;                              // global_addr[31:0]
    g0[3] = (unsigned int)((ga >> 32) & 0x1FFFFFFu)        // global_addr[56:32]
          | (2u << 30);                                    // type = 2 ("image")
    v8i g1;
    g1[0] = (int)(3u << 16);       // wg_mask=0, data_size=3 (8 bytes), no flags
    g1[1] = (int)(4096u << 16);    // tensor_dim0[15:0]=4096 in bits[63:48]
    g1[2] = (int)(1u << 16);       // tensor_dim0[31:16]=0 | tensor_dim1[15:0]=1
    g1[3] = (int)(4096u << 16);    // tensor_dim1[31:16]=0 | tile_dim0=4096
    g1[4] = 1;                     // tile_dim1=1, tile_dim2=0
    g1[5] = 4096;                  // tensor_dim0_stride[31:0]
    g1[6] = 0;                     // stride[47:32]=0 | dim1_stride[15:0]=0
    g1[7] = 0;                     // dim1_stride[47:16]=0
    v4i z4 = {0, 0, 0, 0};
    v8i z8 = {0, 0, 0, 0, 0, 0, 0, 0};
    __builtin_amdgcn_tensor_load_to_lds(g0, g1, z4, z4, z8, 0);
}
#endif

// ---------------------------------------------------------------------------
// K0: convert attn_e_w (f32 row-major [n][k]) -> bf16 row-major
// ---------------------------------------------------------------------------
__global__ __launch_bounds__(256) void k_convert_w(const float* __restrict__ w,
                                                   unsigned int* __restrict__ wbf) {
    int idx = (blockIdx.x * 256 + threadIdx.x) * 8;     // element index, 8 per thread
    const float4* src = (const float4*)(w + idx);
    float4 a = src[0];
    float4 b = src[1];
    wbf[idx / 2 + 0] = pack_bf16(a.x, a.y);
    wbf[idx / 2 + 1] = pack_bf16(a.z, a.w);
    wbf[idx / 2 + 2] = pack_bf16(b.x, b.y);
    wbf[idx / 2 + 3] = pack_bf16(b.z, b.w);
}

// ---------------------------------------------------------------------------
// K1: hproj[b][n] = last_decoder_hidden[b] . attn_h_w[n] + attn_h_b[n]
// ---------------------------------------------------------------------------
__global__ __launch_bounds__(256) void k_hproj(const float* __restrict__ hdd,
                                               const float* __restrict__ w,
                                               const float* __restrict__ bias,
                                               float* __restrict__ hproj) {
    __shared__ __align__(16) float xs[H];
    int b = blockIdx.x, t = threadIdx.x;
    ((float4*)xs)[t] = ((const float4*)(hdd + b * H))[t];
    __syncthreads();
    const float4* xv = (const float4*)xs;
    for (int c = 0; c < 4; ++c) {
        int n = t + 256 * c;
        const float4* wv = (const float4*)(w + n * H);
        float acc = 0.f;
        #pragma unroll 4
        for (int k = 0; k < H / 4; ++k) {
            float4 a = wv[k], x = xv[k];
            acc += a.x * x.x + a.y * x.y + a.z * x.z + a.w * x.w;
        }
        hproj[b * H + n] = acc + bias[n];
    }
}

// ---------------------------------------------------------------------------
// K2: fused big GEMM + sigmoid + energy-dot (the WMMA kernel)
//   E = X * We^T  (X = enc reshaped [65536,1024], We bf16 [n][k])
//   energy[row]  = sum_n sigmoid(E[row,n] + e_b[n] + hproj[row%B, n]) * en_w[n] + en_b
// Workgroup: 256 threads = 8 waves; 128 rows per workgroup (16 per wave).
// LDS: A-slab 128x1024 bf16 (256 KB, staged once) + 2x 32 KB W tiles,
// double-buffered via the Tensor Data Mover (TENSORcnt).
// ---------------------------------------------------------------------------
__global__ __launch_bounds__(256) void k_attn_energy(const float* __restrict__ enc,
                                                     const unsigned int* __restrict__ webf,
                                                     const float* __restrict__ hproj,
                                                     const float* __restrict__ e_b,
                                                     const float* __restrict__ en_w,
                                                     const float* __restrict__ en_b,
                                                     float* __restrict__ energy) {
    extern __shared__ char smem[];
    unsigned int* Abuf = (unsigned int*)smem;      // 128*1024 bf16 = 65536 uints

    const int t       = threadIdx.x;
    const int wave    = t >> 5;
    const int lane    = t & 31;
    const int g       = lane >> 4;        // K half-group selector
    const int ln      = lane & 15;        // M for A-frag / N for B-frag
    const int rowBase = blockIdx.x * 128;

#if HAS_TDM
    // kick off the DMA for W tile 0 while we stage/convert the A slab
    if (wave == 0) tdm_load_w(webf, LDS_A_BYTES);
#endif

    // ---- stage A slab: 128 contiguous rows of enc, f32 -> bf16 into LDS ----
    {
        const float4* src = (const float4*)(enc + (size_t)rowBase * H);
        for (int i = 0; i < 128; ++i) {           // 128 float4 per thread
            int idx = i * 256 + t;                // float4 index in slab
            float4 v = src[idx];
            Abuf[idx * 2 + 0] = pack_bf16(v.x, v.y);
            Abuf[idx * 2 + 1] = pack_bf16(v.z, v.w);
        }
    }

    // hproj row pointers for the 8 accumulator rows this lane owns
    const float* hp[8];
    #pragma unroll
    for (int j = 0; j < 8; ++j) {
        int row = rowBase + 16 * wave + j + 8 * g;
        hp[j] = hproj + (size_t)(row & (B - 1)) * H;   // b = row % B
    }

    float partial[8];
    #pragma unroll
    for (int j = 0; j < 8; ++j) partial[j] = 0.f;

    const int arow = 16 * wave + ln;   // this lane's A-slab row

    for (int nt = 0; nt < H / 16; ++nt) {
        const int nb = nt * 16;

        // barrier: previous compute done -> buffer (nt+1)&1 is reusable,
        // and A-slab staging complete on the first iteration
        __syncthreads();
#if HAS_TDM
        if (wave == 0) {
            if (nt < 63) {
                tdm_load_w(webf + (size_t)(nb + 16) * (H / 2),
                           (unsigned int)(LDS_A_BYTES + ((nt + 1) & 1) * LDS_W_BYTES));
                __builtin_amdgcn_s_wait_tensorcnt(1);   // tile nt landed in LDS
            } else {
                __builtin_amdgcn_s_wait_tensorcnt(0);
            }
        }
#else
        {   // fallback: cooperative copy of W tile nt
            const v4u* src = (const v4u*)(webf + (size_t)nb * (H / 2));
            v4u* dst = (v4u*)(smem + LDS_A_BYTES + (size_t)(nt & 1) * LDS_W_BYTES);
            #pragma unroll
            for (int i = 0; i < 16; ++i) dst[i * 256 + t] = src[i * 256 + t];
        }
#endif
        __syncthreads();   // publish tile nt to all waves

        const char* wtile = smem + LDS_A_BYTES + (size_t)(nt & 1) * LDS_W_BYTES;

        // two independent accumulation chains to pipeline the matrix unit
        v8f acc0 = {0.f, 0.f, 0.f, 0.f, 0.f, 0.f, 0.f, 0.f};
        v8f acc1 = {0.f, 0.f, 0.f, 0.f, 0.f, 0.f, 0.f, 0.f};

        #pragma unroll 4
        for (int kt = 0; kt < H / 64; ++kt) {
            const int kb = kt * 64;
            #pragma unroll
            for (int half = 0; half < 2; ++half) {
                const int kk = kb + 32 * half;
                // A frag: lane (M=ln, half g): 16B at K=kk+8g, 16B at K=kk+16+8g
                const char* abase = (const char*)Abuf + ((size_t)arow * H + kk + 8 * g) * 2;
                v4u a0 = *(const v4u*)(abase);
                v4u a1 = *(const v4u*)(abase + 32);
                // B frag: lane (N=ln, half g): 16B at K=kk+8g, 16B at K=kk+16+8g of W row ln
                const char* bbase = wtile + ((size_t)ln * H + kk + 8 * g) * 2;
                v4u b0 = *(const v4u*)(bbase);
                v4u b1 = *(const v4u*)(bbase + 32);

                v8u au, bu;
                #pragma unroll
                for (int i = 0; i < 4; ++i) { au[i] = a0[i]; au[i + 4] = a1[i]; }
                #pragma unroll
                for (int i = 0; i < 4; ++i) { bu[i] = b0[i]; bu[i + 4] = b1[i]; }
                v16bf af = __builtin_bit_cast(v16bf, au);
                v16bf bf = __builtin_bit_cast(v16bf, bu);

                if (half == 0)
                    acc0 = __builtin_amdgcn_wmma_f32_16x16x32_bf16(
                        false, af, false, bf, (short)0, acc0, false, false);
                else
                    acc1 = __builtin_amdgcn_wmma_f32_16x16x32_bf16(
                        false, af, false, bf, (short)0, acc1, false, false);
            }
        }

        // ---- fused epilogue: sigmoid + dot with en_w, accumulated per M-row ----
        const int   col = nb + ln;          // N column for this lane
        const float eb  = e_b[col];
        const float ew  = en_w[col];
        #pragma unroll
        for (int j = 0; j < 8; ++j) {
            float x = acc0[j] + acc1[j] + eb + hp[j][col];
            partial[j] += ew * sigf(x);
        }
    }

    // ---- reduce partials across the 16 N-lanes of each half; write energy ----
    const float enb = en_b[0];
    #pragma unroll
    for (int j = 0; j < 8; ++j) {
        float v = partial[j];
        v += __shfl_xor(v, 1, 32);
        v += __shfl_xor(v, 2, 32);
        v += __shfl_xor(v, 4, 32);
        v += __shfl_xor(v, 8, 32);
        if (ln == 0) {
            int row = rowBase + 16 * wave + j + 8 * g;
            energy[row] = v + enb;
        }
    }
}

// ---------------------------------------------------------------------------
// K3a: softmax over S per batch column b; writes weights (ws) and attn output
// ---------------------------------------------------------------------------
__global__ __launch_bounds__(256) void k_softmax(const float* __restrict__ energy,
                                                 float* __restrict__ wsoft,
                                                 float* __restrict__ out_attn) {
    __shared__ float red[256];
    int b = blockIdx.x, s = threadIdx.x;
    float e = energy[s * B + b];
    red[s] = e;
    __syncthreads();
    for (int off = 128; off > 0; off >>= 1) {
        if (s < off) red[s] = fmaxf(red[s], red[s + off]);
        __syncthreads();
    }
    float m = red[0];
    __syncthreads();
    float ex = __expf(e - m);
    red[s] = ex;
    __syncthreads();
    for (int off = 128; off > 0; off >>= 1) {
        if (s < off) red[s] += red[s + off];
        __syncthreads();
    }
    float w = ex / red[0];
    wsoft[s * B + b] = w;                 // [s,b] layout for context pass
    out_attn[b * S + s] = w;              // attn_weights [B,1,S]
}

// ---------------------------------------------------------------------------
// K3b: context[b][h] = sum_s w[s,b] * enc[s,b,h]
// ---------------------------------------------------------------------------
__global__ __launch_bounds__(256) void k_context(const float* __restrict__ enc,
                                                 const float* __restrict__ wsoft,
                                                 float* __restrict__ out_ctx) {
    __shared__ float wl[S];
    int b = blockIdx.y, t = threadIdx.x;
    int h = blockIdx.x * 256 + t;
    wl[t] = wsoft[t * B + b];
    __syncthreads();
    float acc = 0.f;
    for (int s = 0; s < S; ++s) {
        if (s + 4 < S)
            __builtin_prefetch(&enc[((size_t)(s + 4) * B + b) * H + h], 0, 0); // global_prefetch_b8
        acc += wl[s] * enc[((size_t)s * B + b) * H + h];
    }
    out_ctx[b * H + h] = acc;
}

// ---------------------------------------------------------------------------
// K4: GRU cell, one block per batch row.
// ---------------------------------------------------------------------------
__global__ __launch_bounds__(256) void k_gru(const float* __restrict__ palette_in,
                                             const float* __restrict__ ctx,
                                             const float* __restrict__ hdd,
                                             const float* __restrict__ w_ih,
                                             const float* __restrict__ w_hh,
                                             const float* __restrict__ b_ih,
                                             const float* __restrict__ b_hh,
                                             float* __restrict__ hidden_out) {
    __shared__ float xg[H + P];
    __shared__ float hg[H];
    __shared__ float gi[3 * H];
    __shared__ float gh[3 * H];
    int b = blockIdx.x, t = threadIdx.x;
    if (t < P) xg[t] = palette_in[b * P + t];
    for (int k = t; k < H; k += 256) {
        xg[P + k] = ctx[b * H + k];
        hg[k]     = hdd[b * H + k];
    }
    __syncthreads();
    for (int c = 0; c < 12; ++c) {
        int j = t + 256 * c;
        const float* wi = w_ih + (size_t)j * (H + P);
        const float* wh = w_hh + (size_t)j * H;
        float ai = b_ih[j], ah = b_hh[j];
        #pragma unroll 4
        for (int k = 0; k < H; ++k) { ai += wi[k] * xg[k]; ah += wh[k] * hg[k]; }
        ai += wi[H] * xg[H] + wi[H + 1] * xg[H + 1] + wi[H + 2] * xg[H + 2];
        gi[j] = ai; gh[j] = ah;
    }
    __syncthreads();
    for (int c = 0; c < 4; ++c) {
        int n = t + 256 * c;
        float r  = sigf(gi[n] + gh[n]);
        float z  = sigf(gi[H + n] + gh[H + n]);
        float ng = tanhf(gi[2 * H + n] + r * gh[2 * H + n]);
        hidden_out[b * H + n] = (1.f - z) * ng + z * hg[n];
    }
}

// ---------------------------------------------------------------------------
// K5a: y = relu(hidden @ out1_w^T + out1_b)
// ---------------------------------------------------------------------------
__global__ __launch_bounds__(256) void k_out1(const float* __restrict__ hidden,
                                              const float* __restrict__ w,
                                              const float* __restrict__ bias,
                                              float* __restrict__ y) {
    __shared__ __align__(16) float xs[H];
    int b = blockIdx.x, t = threadIdx.x;
    ((float4*)xs)[t] = ((const float4*)(hidden + b * H))[t];
    __syncthreads();
    const float4* xv = (const float4*)xs;
    for (int c = 0; c < 4; ++c) {
        int n = t + 256 * c;
        const float4* wv = (const float4*)(w + (size_t)n * H);
        float acc = 0.f;
        #pragma unroll 4
        for (int k = 0; k < H / 4; ++k) {
            float4 a = wv[k], x = xv[k];
            acc += a.x * x.x + a.y * x.y + a.z * x.z + a.w * x.w;
        }
        acc += bias[n];
        y[b * H + n] = fmaxf(acc, 0.f);
    }
}

// ---------------------------------------------------------------------------
// K5b: per-feature batchnorm stats -> scale/shift  (biased variance, as jnp.var)
// ---------------------------------------------------------------------------
__global__ __launch_bounds__(256) void k_bnstats(const float* __restrict__ y,
                                                 const float* __restrict__ bn_w,
                                                 const float* __restrict__ bn_b,
                                                 float* __restrict__ bn) {
    __shared__ float s1[256];
    __shared__ float s2[256];
    int n = blockIdx.x, b = threadIdx.x;
    float v = y[b * H + n];
    s1[b] = v; s2[b] = v * v;
    __syncthreads();
    for (int off = 128; off > 0; off >>= 1) {
        if (b < off) { s1[b] += s1[b + off]; s2[b] += s2[b + off]; }
        __syncthreads();
    }
    if (b == 0) {
        float mu  = s1[0] / (float)B;
        float var = s2[0] / (float)B - mu * mu;
        float a   = bn_w[n] * rsqrtf(var + EPS);
        bn[n]     = a;
        bn[H + n] = bn_b[n] - mu * a;
    }
}

// ---------------------------------------------------------------------------
// K5c: palette[b][p] = sum_n (y[b,n]*a[n]+c[n]) * out2_w[p,n] + out2_b[p]
// ---------------------------------------------------------------------------
__global__ __launch_bounds__(256) void k_palette(const float* __restrict__ y,
                                                 const float* __restrict__ bn,
                                                 const float* __restrict__ w2,
                                                 const float* __restrict__ b2,
                                                 float* __restrict__ out) {
    __shared__ float red[3][256];
    int b = blockIdx.x, t = threadIdx.x;
    float s0 = 0.f, s1 = 0.f, s2 = 0.f;
    for (int c = 0; c < 4; ++c) {
        int n = t + 256 * c;
        float xn = y[b * H + n] * bn[n] + bn[H + n];
        s0 += xn * w2[0 * H + n];
        s1 += xn * w2[1 * H + n];
        s2 += xn * w2[2 * H + n];
    }
    red[0][t] = s0; red[1][t] = s1; red[2][t] = s2;
    __syncthreads();
    for (int off = 128; off > 0; off >>= 1) {
        if (t < off) {
            red[0][t] += red[0][t + off];
            red[1][t] += red[1][t + off];
            red[2][t] += red[2][t + off];
        }
        __syncthreads();
    }
    if (t < P) out[b * P + t] = red[t][0] + b2[t];
}

// ---------------------------------------------------------------------------
// Host launcher
// ---------------------------------------------------------------------------
extern "C" void kernel_launch(void* const* d_in, const int* in_sizes, int n_in,
                              void* d_out, int out_size, void* d_ws, size_t ws_size,
                              hipStream_t stream) {
    const float* last_palette = (const float*)d_in[0];
    const float* hdd          = (const float*)d_in[1];
    const float* enc          = (const float*)d_in[2];
    const float* attn_e_w     = (const float*)d_in[3];
    const float* attn_e_b     = (const float*)d_in[4];
    const float* attn_h_w     = (const float*)d_in[5];
    const float* attn_h_b     = (const float*)d_in[6];
    const float* attn_en_w    = (const float*)d_in[7];
    const float* attn_en_b    = (const float*)d_in[8];
    const float* gru_w_ih     = (const float*)d_in[9];
    const float* gru_w_hh     = (const float*)d_in[10];
    const float* gru_b_ih     = (const float*)d_in[11];
    const float* gru_b_hh     = (const float*)d_in[12];
    const float* out1_w       = (const float*)d_in[13];
    const float* out1_b       = (const float*)d_in[14];
    const float* bn_w         = (const float*)d_in[15];
    const float* bn_b         = (const float*)d_in[16];
    const float* out2_w       = (const float*)d_in[17];
    const float* out2_b       = (const float*)d_in[18];
    // d_in[19]=each_input_size, d_in[20]=i  (reference path i==1 is implemented)

    char* ws = (char*)d_ws;
    unsigned int* ws_we   = (unsigned int*)(ws + WS_WE);
    float* ws_hproj       = (float*)(ws + WS_HPROJ);
    float* ws_energy      = (float*)(ws + WS_ENERGY);
    float* ws_wsoft       = (float*)(ws + WS_WSOFT);
    float* ws_y           = (float*)(ws + WS_Y);
    float* ws_bn          = (float*)(ws + WS_BN);

    float* out = (float*)d_out;

    // 0) weight f32 -> bf16
    k_convert_w<<<512, 256, 0, stream>>>(attn_e_w, ws_we);
    // 1) h-projection
    k_hproj<<<B, 256, 0, stream>>>(hdd, attn_h_w, attn_h_b, ws_hproj);
    // 2) fused bf16-WMMA GEMM + sigmoid + energy dot (320 KB dynamic LDS,
    //    TDM double-buffered W tiles)
    k_attn_energy<<<SB / 128, 256, LDS_TOTAL, stream>>>(
        enc, ws_we, ws_hproj, attn_e_b, attn_en_w, attn_en_b, ws_energy);
    // 3) softmax over S (+ attn_weights output) and context
    k_softmax<<<B, S, 0, stream>>>(ws_energy, ws_wsoft, out + OUT_ATTN);
    k_context<<<dim3(H / 256, B), 256, 0, stream>>>(enc, ws_wsoft, out + OUT_CONTEXT);
    // 4) GRU cell
    k_gru<<<B, 256, 0, stream>>>(last_palette, out + OUT_CONTEXT, hdd,
                                 gru_w_ih, gru_w_hh, gru_b_ih, gru_b_hh,
                                 out + OUT_HIDDEN);
    // 5) output head: Linear -> ReLU -> BatchNorm -> Linear
    k_out1<<<B, 256, 0, stream>>>(out + OUT_HIDDEN, out1_w, out1_b, ws_y);
    k_bnstats<<<H, B, 0, stream>>>(ws_y, bn_w, bn_b, ws_bn);
    k_palette<<<B, 256, 0, stream>>>(ws_y, ws_bn, out2_w, out2_b, out + OUT_PALETTE);
}
